// MotifContrastiveModel_80418967650813
// MI455X (gfx1250) — matrix operand order repaired and checked by reading, not compiled, not run on hardware
//
#include <hip/hip_runtime.h>
#include <hip/hip_bf16.h>
#include <cmath>

#define EMB 300
#define NLAYERS 5
#define BN_EPS 1e-5f

typedef __attribute__((ext_vector_type(16))) __bf16 v16bf;
typedef __attribute__((ext_vector_type(8)))  __bf16 v8bf;
typedef __attribute__((ext_vector_type(8)))  float  v8f;

static __device__ __forceinline__ v8bf load8bf(const __bf16* p) {
    union { uint4 u; v8bf b; } t;
    t.u = *reinterpret_cast<const uint4*>(p);
    return t.b;
}

static __device__ __forceinline__ v16bf comb(v8bf lo, v8bf hi) {
    v16bf r;
#pragma unroll
    for (int i = 0; i < 8; ++i) { r[i] = lo[i]; r[i + 8] = hi[i]; }
    return r;
}

// ---------------------------------------------------------------------------
// Register-blocked bf16 WMMA GEMM:  C[M,Ncols] = A[M,Kpad] * Bt^T
//   (Bt stored transposed, [n][Kpad], zero-padded to 64*ngrpN rows)
// One wave computes a 32x64 macro-tile: 2 M-tiles x 4 N-tiles, 8 independent
// f32 accumulators. Per K-step: 2 A-frag + 4 B-frag b128 loads -> 8 WMMAs.
// M must be a multiple of 32 (N=100000, B=2048 both are). Kpad % 32 == 0.
// ---------------------------------------------------------------------------
__global__ __launch_bounds__(256) void k_gemm_wmma(
    const __bf16* __restrict__ A, const __bf16* __restrict__ Bt,
    const float* __restrict__ bias, float* __restrict__ Cf,
    __bf16* __restrict__ Cb, int M, int Ncols, int Kpad,
    int ldc, int ldcb, float scale, int relu, int ngrpN)
{
    int wave = (int)((blockIdx.x * blockDim.x + threadIdx.x) >> 5);
    int lane = threadIdx.x & 31;
    int gN = wave % ngrpN;
    int gM = wave / ngrpN;
    if (gM * 32 >= M) return;                    // wave-uniform exit, EXEC stays full

    int half = lane >> 4;                        // ISA 7.12.2 half-wave split
    int l16  = lane & 15;
    const __bf16* arow0 = A + (size_t)(gM * 32 + l16) * Kpad;
    const __bf16* arow1 = arow0 + (size_t)16 * Kpad;
    const __bf16* brow0 = Bt + (size_t)(gN * 64 + l16) * Kpad;
    const __bf16* brow1 = brow0 + (size_t)16 * Kpad;
    const __bf16* brow2 = brow0 + (size_t)32 * Kpad;
    const __bf16* brow3 = brow0 + (size_t)48 * Kpad;
    const int aoff = half * 16;                  // A: lanes>=16 hold K+16..31
    const int boff = half * 16;                  // B: lanes>=16 hold K+16..31

    v8f c[2][4];
#pragma unroll
    for (int i = 0; i < 2; ++i)
#pragma unroll
        for (int j = 0; j < 4; ++j)
            c[i][j] = (v8f){0.f, 0.f, 0.f, 0.f, 0.f, 0.f, 0.f, 0.f};

    for (int kb = 0; kb < Kpad; kb += 32) {
        v16bf a0 = comb(load8bf(arow0 + kb + aoff), load8bf(arow0 + kb + aoff + 8));
        v16bf a1 = comb(load8bf(arow1 + kb + aoff), load8bf(arow1 + kb + aoff + 8));
        v16bf b0 = comb(load8bf(brow0 + kb + boff), load8bf(brow0 + kb + boff + 8));
        v16bf b1 = comb(load8bf(brow1 + kb + boff), load8bf(brow1 + kb + boff + 8));
        v16bf b2 = comb(load8bf(brow2 + kb + boff), load8bf(brow2 + kb + boff + 8));
        v16bf b3 = comb(load8bf(brow3 + kb + boff), load8bf(brow3 + kb + boff + 8));
        c[0][0] = __builtin_amdgcn_wmma_f32_16x16x32_bf16(false, a0, false, b0, (short)0, c[0][0], false, false);
        c[0][1] = __builtin_amdgcn_wmma_f32_16x16x32_bf16(false, a0, false, b1, (short)0, c[0][1], false, false);
        c[0][2] = __builtin_amdgcn_wmma_f32_16x16x32_bf16(false, a0, false, b2, (short)0, c[0][2], false, false);
        c[0][3] = __builtin_amdgcn_wmma_f32_16x16x32_bf16(false, a0, false, b3, (short)0, c[0][3], false, false);
        c[1][0] = __builtin_amdgcn_wmma_f32_16x16x32_bf16(false, a1, false, b0, (short)0, c[1][0], false, false);
        c[1][1] = __builtin_amdgcn_wmma_f32_16x16x32_bf16(false, a1, false, b1, (short)0, c[1][1], false, false);
        c[1][2] = __builtin_amdgcn_wmma_f32_16x16x32_bf16(false, a1, false, b2, (short)0, c[1][2], false, false);
        c[1][3] = __builtin_amdgcn_wmma_f32_16x16x32_bf16(false, a1, false, b3, (short)0, c[1][3], false, false);
    }

#pragma unroll
    for (int j = 0; j < 4; ++j) {
        int n = gN * 64 + j * 16 + l16;
        if (n < Ncols) {
            float bb = bias ? bias[n] : 0.0f;
#pragma unroll
            for (int i = 0; i < 2; ++i) {
                int mbase = gM * 32 + i * 16 + half * 8;   // C VGPR r -> row half*8+r
#pragma unroll
                for (int r = 0; r < 8; ++r) {
                    float v = c[i][j][r] * scale + bb;
                    if (relu) v = fmaxf(v, 0.0f);
                    if (Cf) Cf[(size_t)(mbase + r) * ldc  + n] = v;
                    if (Cb) Cb[(size_t)(mbase + r) * ldcb + n] = (__bf16)v;
                }
            }
        }
    }
}

// ---------------------------------------------------------------------------
// Elementwise / graph kernels
// ---------------------------------------------------------------------------
__global__ void k_embed(const int* __restrict__ x, const float* __restrict__ a1,
                        const float* __restrict__ a2, float* __restrict__ h, int N)
{
    long long idx = (long long)blockIdx.x * blockDim.x + threadIdx.x;
    if (idx >= (long long)N * EMB) return;
    int i = (int)(idx / EMB), d = (int)(idx % EMB);
    h[(size_t)i * EMB + d] = a1[(size_t)x[2 * i] * EMB + d] +
                             a2[(size_t)x[2 * i + 1] * EMB + d];
}

// aggr = h + (bond1[l][4] + bond2[l][0]);  aggr stored with ld=304
__global__ void k_aggr_init(const float* __restrict__ h, const float* __restrict__ s1,
                            const float* __restrict__ s2, float* __restrict__ aggr, int N)
{
    long long idx = (long long)blockIdx.x * blockDim.x + threadIdx.x;
    if (idx >= (long long)N * EMB) return;
    int i = (int)(idx / EMB), d = (int)(idx % EMB);
    aggr[(size_t)i * 304 + d] = h[(size_t)i * EMB + d] + s1[d] + s2[d];
}

// one block per edge: aggr[dst] += h[src] + e(edge_attr)
__global__ void k_edge_msg(const int* __restrict__ ei, const int* __restrict__ ea,
                           const float* __restrict__ h, const float* __restrict__ be1,
                           const float* __restrict__ be2, float* __restrict__ aggr, int E)
{
    int e = blockIdx.x;
    int src = ei[e], dst = ei[E + e];
    int a0 = ea[(size_t)e * 2], a1 = ea[(size_t)e * 2 + 1];
    const float* hs = h   + (size_t)src * EMB;
    const float* e1 = be1 + (size_t)a0 * EMB;
    const float* e2 = be2 + (size_t)a1 * EMB;
    float* out = aggr + (size_t)dst * 304;
    for (int d = threadIdx.x; d < EMB; d += blockDim.x)
        atomicAdd(&out[d], hs[d] + e1[d] + e2[d]);
}

__global__ void k_f32_to_bf16_pad(const float* __restrict__ in, int ldin, int validc,
                                  __bf16* __restrict__ out, int ldout, int rows)
{
    long long idx = (long long)blockIdx.x * blockDim.x + threadIdx.x;
    if (idx >= (long long)rows * ldout) return;
    int r = (int)(idx / ldout), c = (int)(idx % ldout);
    out[idx] = (c < validc) ? (__bf16)in[(size_t)r * ldin + c] : (__bf16)0.0f;
}

// W[K,Nc] f32  ->  Wt[Nrp][Kpad] bf16 (transposed, zero-padded)
__global__ void k_transpose_w(const float* __restrict__ W, int K, int Nc,
                              __bf16* __restrict__ Wt, int Kpad, int Nrp)
{
    long long idx = (long long)blockIdx.x * blockDim.x + threadIdx.x;
    if (idx >= (long long)Nrp * Kpad) return;
    int n = (int)(idx / Kpad), k = (int)(idx % Kpad);
    Wt[idx] = (n < Nc && k < K) ? (__bf16)W[(size_t)k * Nc + n] : (__bf16)0.0f;
}

__global__ void k_pad_zero_bf16(__bf16* __restrict__ p, int ld, int c0, int c1, int rows)
{
    int w = c1 - c0;
    long long idx = (long long)blockIdx.x * blockDim.x + threadIdx.x;
    if (idx >= (long long)rows * w) return;
    int r = (int)(idx / w), c = c0 + (int)(idx % w);
    p[(size_t)r * ld + c] = (__bf16)0.0f;
}

__global__ void k_zero_f32(float* __restrict__ p, long long n)
{
    long long idx = (long long)blockIdx.x * blockDim.x + threadIdx.x;
    if (idx < n) p[idx] = 0.0f;
}

__global__ void k_zero2(float* __restrict__ a, float* __restrict__ b, int n)
{
    int i = threadIdx.x;
    if (i < n) { a[i] = 0.0f; b[i] = 0.0f; }
}

// per-block partial column sums via LDS f32 atomics, then global atomics
__global__ void k_bn_stats(const float* __restrict__ h2, int M, int ld,
                           float* __restrict__ sums, float* __restrict__ sumsq)
{
    __shared__ float ss[EMB];
    __shared__ float sq[EMB];
    int tid = threadIdx.x;
    for (int c = tid; c < EMB; c += blockDim.x) { ss[c] = 0.f; sq[c] = 0.f; }
    __syncthreads();
    int r0 = blockIdx.x * 32;
    for (int t = tid; t < 32 * EMB; t += blockDim.x) {
        int r = r0 + t / EMB;
        if (r < M) {
            int c = t % EMB;
            float v = h2[(size_t)r * ld + c];
            atomicAdd(&ss[c], v);
            atomicAdd(&sq[c], v * v);
        }
    }
    __syncthreads();
    for (int c = tid; c < EMB; c += blockDim.x) {
        atomicAdd(&sums[c], ss[c]);
        atomicAdd(&sumsq[c], sq[c]);
    }
}

__global__ void k_bn_finalize(const float* __restrict__ sums, const float* __restrict__ sumsq,
                              int M, float* __restrict__ mean, float* __restrict__ rstd)
{
    int c = threadIdx.x;
    if (c < EMB) {
        float m = sums[c] / (float)M;
        float v = sumsq[c] / (float)M - m * m;
        mean[c] = m;
        rstd[c] = rsqrtf(v + BN_EPS);
    }
}

__global__ void k_bn_apply(const float* __restrict__ h2, int ld,
                           const float* __restrict__ mean, const float* __restrict__ rstd,
                           const float* __restrict__ g, const float* __restrict__ b,
                           float* __restrict__ h, int N, int relu)
{
    long long idx = (long long)blockIdx.x * blockDim.x + threadIdx.x;
    if (idx >= (long long)N * EMB) return;
    int i = (int)(idx / EMB), c = (int)(idx % EMB);
    float v = g[c] * (h2[(size_t)i * ld + c] - mean[c]) * rstd[c] + b[c];
    if (relu) v = fmaxf(v, 0.0f);
    h[(size_t)i * EMB + c] = v;
}

__global__ void k_pool_acc(const float* __restrict__ h, const int* __restrict__ batch,
                           float* __restrict__ sums, float* __restrict__ cnts, int N)
{
    long long idx = (long long)blockIdx.x * blockDim.x + threadIdx.x;
    if (idx >= (long long)N * EMB) return;
    int i = (int)(idx / EMB), d = (int)(idx % EMB);
    int b = batch[i];
    atomicAdd(&sums[(size_t)b * EMB + d], h[(size_t)i * EMB + d]);
    if (d == 0) atomicAdd(&cnts[b], 1.0f);
}

__global__ void k_pool_div(float* __restrict__ sums, const float* __restrict__ cnts, int B)
{
    long long idx = (long long)blockIdx.x * blockDim.x + threadIdx.x;
    if (idx >= (long long)B * EMB) return;
    sums[idx] /= fmaxf(cnts[idx / EMB], 1.0f);
}

// row-wise L2 normalize + convert to bf16 with zero pad to ld 320
__global__ void k_l2norm_bf16(const float* __restrict__ in, int ld,
                              __bf16* __restrict__ out, int rows)
{
    __shared__ float red[128];
    int row = blockIdx.x, tid = threadIdx.x;
    float ss = 0.f;
    for (int c = tid; c < EMB; c += 128) {
        float v = in[(size_t)row * ld + c];
        ss += v * v;
    }
    red[tid] = ss;
    __syncthreads();
    for (int s = 64; s > 0; s >>= 1) {
        if (tid < s) red[tid] += red[tid + s];
        __syncthreads();
    }
    float rinv = 1.0f / fmaxf(sqrtf(red[0]), 1e-12f);
    for (int c = tid; c < 320; c += 128)
        out[(size_t)row * 320 + c] =
            (c < EMB) ? (__bf16)(in[(size_t)row * ld + c] * rinv) : (__bf16)0.0f;
}

// ---------------------------------------------------------------------------
static inline int cdiv(long long a, long long b) { return (int)((a + b - 1) / b); }

extern "C" void kernel_launch(void* const* d_in, const int* in_sizes, int n_in,
                              void* d_out, int out_size, void* d_ws, size_t ws_size,
                              hipStream_t stream)
{
    const int*  x[2]     = {(const int*)d_in[0], (const int*)d_in[4]};
    const int*  ei[2]    = {(const int*)d_in[1], (const int*)d_in[5]};
    const int*  ea[2]    = {(const int*)d_in[2], (const int*)d_in[6]};
    const int*  batch[2] = {(const int*)d_in[3], (const int*)d_in[7]};
    const float* atom1 = (const float*)d_in[8];
    const float* atom2 = (const float*)d_in[9];
    const float* bond1 = (const float*)d_in[10];
    const float* bond2 = (const float*)d_in[11];
    const float* W1 = (const float*)d_in[12];
    const float* b1 = (const float*)d_in[13];
    const float* W2 = (const float*)d_in[14];
    const float* b2 = (const float*)d_in[15];
    const float* bng = (const float*)d_in[16];
    const float* bnb = (const float*)d_in[17];
    const float* projW = (const float*)d_in[18];
    const float* projb = (const float*)d_in[19];

    const int N = in_sizes[3];                 // 100000 (multiple of 32)
    const int E = in_sizes[2] / 2;             // 250000
    const int B = (int)(sqrt((double)out_size) + 0.5);   // 2048

    // ---- workspace carve-up (256B aligned) --------------------------------
    char* ws = (char*)d_ws;
    size_t off = 0;
    auto alloc = [&](size_t bytes) -> void* {
        void* p = ws + off;
        off += (bytes + 255) & ~(size_t)255;
        return p;
    };
    float*  h      = (float*) alloc((size_t)N * EMB * 4);
    float*  aggr   = (float*) alloc((size_t)N * 304 * 4);   // also h2 (GEMM2 out)
    __bf16* aggrb  = (__bf16*)alloc((size_t)N * 320 * 2);
    __bf16* h2a    = (__bf16*)alloc((size_t)N * 640 * 2);
    __bf16* W1t    = (__bf16*)alloc((size_t)640 * 320 * 2); // 10 N-groups * 64 rows
    __bf16* W2t    = (__bf16*)alloc((size_t)320 * 640 * 2); // 5 N-groups * 64 rows
    __bf16* projt  = (__bf16*)alloc((size_t)320 * 320 * 2); // 5 N-groups * 64 rows
    float*  bnsum  = (float*) alloc(EMB * 4);
    float*  bnsq   = (float*) alloc(EMB * 4);
    float*  bnmean = (float*) alloc(EMB * 4);
    float*  bnrstd = (float*) alloc(EMB * 4);
    float*  pooled[2] = {(float*)alloc((size_t)B * EMB * 4),
                         (float*)alloc((size_t)B * EMB * 4)};
    float*  cnts    = (float*) alloc((size_t)B * 4);
    float*  projout = (float*) alloc((size_t)B * 304 * 4);
    __bf16* pooledb = (__bf16*)alloc((size_t)B * 320 * 2);
    __bf16* featb[2] = {(__bf16*)alloc((size_t)B * 320 * 2),
                        (__bf16*)alloc((size_t)B * 320 * 2)};
    (void)ws_size; (void)n_in;

    const long long NE = (long long)N * EMB;
    const int mg = N / 32;                     // 3125 row macro-tiles

    // GEMM-2 K-pad columns (600..639) of h2a are never written by the guarded
    // store: zero them once.
    k_pad_zero_bf16<<<cdiv((long long)N * 40, 256), 256, 0, stream>>>(h2a, 640, 600, 640, N);

    for (int v = 0; v < 2; ++v) {
        // node embedding
        k_embed<<<cdiv(NE, 256), 256, 0, stream>>>(x[v], atom1, atom2, h, N);

        for (int l = 0; l < NLAYERS; ++l) {
            const float* be1 = bond1 + (size_t)l * 6 * EMB;
            const float* be2 = bond2 + (size_t)l * 3 * EMB;
            // aggr = h + selfloop(bond1[l][4] + bond2[l][0])
            k_aggr_init<<<cdiv(NE, 256), 256, 0, stream>>>(h, be1 + 4 * EMB, be2, aggr, N);
            // scatter-add messages
            k_edge_msg<<<E, 128, 0, stream>>>(ei[v], ea[v], h, be1, be2, aggr, E);
            // aggr -> bf16 [N,320] zero-padded
            k_f32_to_bf16_pad<<<cdiv((long long)N * 320, 256), 256, 0, stream>>>(
                aggr, 304, EMB, aggrb, 320, N);

            // MLP1: relu(aggr @ W1 + b1) -> h2a bf16 [N,640]; 10 N-groups
            k_transpose_w<<<cdiv(640 * 320, 256), 256, 0, stream>>>(
                W1 + (size_t)l * EMB * 600, EMB, 600, W1t, 320, 640);
            k_gemm_wmma<<<cdiv((long long)mg * 10 * 32, 256), 256, 0, stream>>>(
                aggrb, W1t, b1 + (size_t)l * 600, nullptr, h2a,
                N, 600, 320, 0, 640, 1.0f, 1, 10);

            // MLP2: h2a @ W2 + b2 -> h2 f32 (reuse aggr, ld 304); 5 N-groups
            k_transpose_w<<<cdiv(320 * 640, 256), 256, 0, stream>>>(
                W2 + (size_t)l * 600 * EMB, 600, EMB, W2t, 640, 320);
            k_gemm_wmma<<<cdiv((long long)mg * 5 * 32, 256), 256, 0, stream>>>(
                h2a, W2t, b2 + (size_t)l * EMB, aggr, nullptr,
                N, EMB, 640, 304, 0, 1.0f, 0, 5);

            // BatchNorm (batch statistics) + relu (except last layer)
            k_zero2<<<1, 320, 0, stream>>>(bnsum, bnsq, EMB);
            k_bn_stats<<<cdiv(N, 32), 256, 0, stream>>>(aggr, N, 304, bnsum, bnsq);
            k_bn_finalize<<<1, 320, 0, stream>>>(bnsum, bnsq, N, bnmean, bnrstd);
            k_bn_apply<<<cdiv(NE, 256), 256, 0, stream>>>(
                aggr, 304, bnmean, bnrstd, bng + (size_t)l * EMB, bnb + (size_t)l * EMB,
                h, N, (l < NLAYERS - 1) ? 1 : 0);
        }

        // mean pool per graph
        k_zero_f32<<<cdiv((long long)B * EMB, 256), 256, 0, stream>>>(pooled[v], (long long)B * EMB);
        k_zero_f32<<<cdiv(B, 256), 256, 0, stream>>>(cnts, B);
        k_pool_acc<<<cdiv(NE, 256), 256, 0, stream>>>(h, batch[v], pooled[v], cnts, N);
        k_pool_div<<<cdiv((long long)B * EMB, 256), 256, 0, stream>>>(pooled[v], cnts, B);
    }

    // feat0 = l2norm(pooled0)
    k_l2norm_bf16<<<B, 128, 0, stream>>>(pooled[0], EMB, featb[0], B);

    // feat1 = l2norm(pooled1 @ proj_W + proj_b); 5 N-groups
    k_f32_to_bf16_pad<<<cdiv((long long)B * 320, 256), 256, 0, stream>>>(
        pooled[1], EMB, EMB, pooledb, 320, B);
    k_transpose_w<<<cdiv(320 * 320, 256), 256, 0, stream>>>(projW, EMB, EMB, projt, 320, 320);
    k_gemm_wmma<<<cdiv((long long)(B / 32) * 5 * 32, 256), 256, 0, stream>>>(
        pooledb, projt, projb, projout, nullptr, B, EMB, 320, 304, 0, 1.0f, 0, 5);
    k_l2norm_bf16<<<B, 128, 0, stream>>>(projout, 304, featb[1], B);

    // logits = feat0 @ feat1^T / TEMP  — featb[1] row-major IS the Bt layout,
    // B/64 = 32 N-groups (2048 rows exactly, no padding needed)
    k_gemm_wmma<<<cdiv((long long)(B / 32) * (B / 64) * 32, 256), 256, 0, stream>>>(
        featb[0], featb[1], nullptr, (float*)d_out, nullptr,
        B, B, 320, B, 0, 1.0f / 0.04f, 0, B / 64);
}
